// SelfAttentionBlock_2723009265925
// MI455X (gfx1250) — matrix operand order repaired and checked by reading, not compiled, not run on hardware
//
#include <hip/hip_runtime.h>
#include <cstdint>

// ---------------------------------------------------------------------------
// Self-attention block for MI455X (gfx1250), bf16 WMMA flash-attention.
// B=4, C=256, H=W=64 -> N=4096, CQ=32.
// ---------------------------------------------------------------------------

typedef __attribute__((ext_vector_type(16))) __bf16          v16bf;
typedef __attribute__((ext_vector_type(8)))  float           v8f;
typedef __attribute__((ext_vector_type(8)))  unsigned short  ushort8;

union Frag {
  v16bf   v;
  ushort8 u8[2];
};

#define LOG2E 1.44269504088896340736f

#if __has_builtin(__builtin_amdgcn_exp2f)
#define EXP2F(x) __builtin_amdgcn_exp2f(x)
#else
#define EXP2F(x) exp2f(x)
#endif

static __device__ __forceinline__ unsigned short f2bf(float x) {
  unsigned u = __builtin_bit_cast(unsigned, x);
  unsigned r = (u + 0x7FFFu + ((u >> 16) & 1u)) >> 16;   // RNE
  return (unsigned short)r;
}

static __device__ __forceinline__ v8f vzero8() {
  v8f z;
#pragma unroll
  for (int e = 0; e < 8; ++e) z[e] = 0.0f;
  return z;
}

// ---------------------------------------------------------------------------
// Kernel 0: pack Wq|Wk|Wv -> bf16 Wall[320][256], bq|bk|bv -> ball[320]
// ---------------------------------------------------------------------------
__global__ __launch_bounds__(256) void k_prep(
    const float* __restrict__ Wq, const float* __restrict__ Wk,
    const float* __restrict__ Wv, const float* __restrict__ bq,
    const float* __restrict__ bk, const float* __restrict__ bv,
    unsigned short* __restrict__ Wall, float* __restrict__ ball) {
  int idx = blockIdx.x * 256 + threadIdx.x;
  if (idx < 320 * 256) {
    int o = idx >> 8, c = idx & 255;
    float v = (o < 32) ? Wq[o * 256 + c]
            : (o < 64) ? Wk[(o - 32) * 256 + c]
                       : Wv[(o - 64) * 256 + c];
    Wall[idx] = f2bf(v);
  }
  if (idx < 320) {
    ball[idx] = (idx < 32) ? bq[idx] : (idx < 64) ? bk[idx - 32] : bv[idx - 64];
  }
}

// ---------------------------------------------------------------------------
// Kernel 1: fused Q/K/V 1x1-conv projection via bf16 WMMA.
// Out[n][o] = sum_c x[b][c][n] * Wall[o][c] + ball[o]
// Q,K stored bf16 [B][4096][32] row-major.
// V stored bf16 in WMMA-B-fragment tile layout:
//   Vg[b][n>>5][c>>4][(n&31)>>4)*16 + (c&15)][n&15]
// Grid (64, 4): 64-pixel tile per block, 8 waves = 4 n-subtiles x 2 o-halves.
// ---------------------------------------------------------------------------
__global__ __launch_bounds__(256) void k_proj(
    const float* __restrict__ x, const unsigned short* __restrict__ Wall,
    const float* __restrict__ ball, unsigned short* __restrict__ Qws,
    unsigned short* __restrict__ Kws, unsigned short* __restrict__ Vg) {
  const int b    = blockIdx.y;
  const int n0   = blockIdx.x * 64;
  const int tid  = threadIdx.x;
  const int lane = tid & 31, wave = tid >> 5;
  const int wi = wave & 3, wo = wave >> 2;
  const int lhi = lane >> 4, ln = lane & 15;

  __shared__ unsigned short AF[4][32][16];  // A fragments, 4 KB

  v8f acc[10];
#pragma unroll
  for (int t = 0; t < 10; ++t) acc[t] = vzero8();

  for (int kc = 0; kc < 256; kc += 32) {
    __syncthreads();
    {  // stage x[kc..kc+31][n0..n0+63] into A-fragment layout (f32 -> bf16)
      const int cc = tid >> 3;
      const int nb = (tid & 7) * 8;
      const float* xr = x + (size_t)(b * 256 + kc + cc) * 4096 + n0 + nb;
      float4 x0 = *(const float4*)xr;
      float4 x1 = *(const float4*)(xr + 4);
      const int hi = cc >> 4;
      const int ib = cc & 15;
      float vals[8] = {x0.x, x0.y, x0.z, x0.w, x1.x, x1.y, x1.z, x1.w};
#pragma unroll
      for (int q = 0; q < 8; ++q) {
        int n = nb + q;
        AF[n >> 4][(n & 15) + 16 * hi][ib] = f2bf(vals[q]);
      }
    }
    __syncthreads();

    Frag a;
    a.u8[0] = *(const ushort8*)&AF[wi][lane][0];
    a.u8[1] = *(const ushort8*)&AF[wi][lane][8];
    const int kb = lhi * 16;
#pragma unroll
    for (int t = 0; t < 10; ++t) {
      const int o = wo * 160 + t * 16 + ln;
      const unsigned short* wr = Wall + (size_t)o * 256 + kc + kb;
      Frag bf;
      bf.u8[0] = *(const ushort8*)wr;
      bf.u8[1] = *(const ushort8*)(wr + 8);
      acc[t] = __builtin_amdgcn_wmma_f32_16x16x32_bf16(
          false, a.v, false, bf.v, (short)0, acc[t], false, false);
    }
  }

  // epilogue: bias + scatter to Q/K/V layouts
#pragma unroll
  for (int t = 0; t < 10; ++t) {
    const int o = wo * 160 + t * 16 + ln;
    const float bias = ball[o];
#pragma unroll
    for (int r = 0; r < 8; ++r) {
      const int row = n0 + wi * 16 + r + 8 * lhi;  // pixel index n
      unsigned short us = f2bf(acc[t][r] + bias);
      if (o < 32) {
        Qws[(size_t)(b * 4096 + row) * 32 + o] = us;
      } else if (o < 64) {
        Kws[(size_t)(b * 4096 + row) * 32 + (o - 32)] = us;
      } else {
        const int c = o - 64;
        const int jblk = row >> 5, jj = row & 31;
        Vg[(size_t)(b * 128 + jblk) * 8192 + (c >> 4) * 512 +
           ((jj >> 4) * 16 + (c & 15)) * 16 + (jj & 15)] = us;
      }
    }
  }
}

// ---------------------------------------------------------------------------
// Kernel 2: flash attention + residual.
// Grid (64, 4): one block = (b, 64 query rows). 8 waves: (wi 0..3) x (wc 0..1),
// wave owns rows i0+wi*16..+16, channels wc*128..+128 (8 f32 accum tiles).
// Streams V in 32-pixel tiles via async global->LDS with double buffering.
// ---------------------------------------------------------------------------
__global__ __launch_bounds__(256) void k_attn(
    const float* __restrict__ x, const unsigned short* __restrict__ Qws,
    const unsigned short* __restrict__ Kws, const unsigned short* __restrict__ Vg,
    float* __restrict__ out) {
  const int b    = blockIdx.y;
  const int i0   = blockIdx.x * 64;
  const int tid  = threadIdx.x;
  const int lane = tid & 31, wave = tid >> 5;
  const int wi = wave & 3, wc = wave >> 2;
  const int lhi = lane >> 4, ln = lane & 15;

  __shared__ __align__(16) char smem[40960];
  unsigned short* VF = (unsigned short*)smem;             // [2][16][32][16] 32KB
  unsigned short* PL = (unsigned short*)(smem + 32768);   // [8][16][32]      8KB

  // persistent Q A-fragment (rows i0+wi*16.., K-dim = CQ = 32)
  Frag qf;
  {
    const unsigned short* qr =
        Qws + (size_t)(b * 4096 + i0 + wi * 16 + ln) * 32 + lhi * 16;
    qf.u8[0] = *(const ushort8*)qr;          // K = 16*lhi + 0..7
    qf.u8[1] = *(const ushort8*)(qr + 8);    // K = 16*lhi + 8..15
  }

  v8f O[8];
  float mr[8], lr[8];
#pragma unroll
  for (int t = 0; t < 8; ++t) O[t] = vzero8();
#pragma unroll
  for (int r = 0; r < 8; ++r) { mr[r] = -3.0e38f; lr[r] = 0.0f; }

  const unsigned short* Vbase = Vg + (size_t)b * 128 * 8192;

  auto stage = [&](int buf, int jn) {
    const char* g = (const char*)(Vbase + (size_t)(jn >> 5) * 8192);
    unsigned lbase = (unsigned)(unsigned long long)(const void*)(VF + buf * 8192);
#pragma unroll
    for (int q = 0; q < 4; ++q) {
      unsigned loff = lbase + (unsigned)(tid * 16 + q * 4096);
      const char* gp = g + tid * 16 + q * 4096;
      asm volatile("global_load_async_to_lds_b128 %0, %1, off"
                   :: "v"(loff), "v"(gp) : "memory");
    }
  };

  stage(0, 0);
  asm volatile("s_wait_asynccnt 0" ::: "memory");
  __syncthreads();

  for (int j0 = 0; j0 < 4096; j0 += 32) {
    const int buf = (j0 >> 5) & 1;
    if (j0 + 32 < 4096) stage(buf ^ 1, j0 + 32);

    // ---- S = Q . K^T for 32 keys (two 16x16 tiles) ----
    Frag kf0, kf1;
    {
      const unsigned short* kr =
          Kws + (size_t)(b * 4096 + j0 + ln) * 32 + lhi * 16;
      kf0.u8[0] = *(const ushort8*)kr;
      kf0.u8[1] = *(const ushort8*)(kr + 8);
      const unsigned short* kr1 = kr + 16 * 32;
      kf1.u8[0] = *(const ushort8*)kr1;
      kf1.u8[1] = *(const ushort8*)(kr1 + 8);
    }
    v8f S0 = __builtin_amdgcn_wmma_f32_16x16x32_bf16(
        false, qf.v, false, kf0.v, (short)0, vzero8(), false, false);
    v8f S1 = __builtin_amdgcn_wmma_f32_16x16x32_bf16(
        false, qf.v, false, kf1.v, (short)0, vzero8(), false, false);

    // ---- online softmax (row = r + 8*lhi; reduce across 16-lane half) ----
    float rmax[8], rsum[8], p0[8], p1[8], alpha[8];
#pragma unroll
    for (int r = 0; r < 8; ++r) rmax[r] = fmaxf(S0[r], S1[r]);
#pragma unroll
    for (int m = 1; m < 16; m <<= 1)
#pragma unroll
      for (int r = 0; r < 8; ++r)
        rmax[r] = fmaxf(rmax[r], __shfl_xor(rmax[r], m, 32));
#pragma unroll
    for (int r = 0; r < 8; ++r) {
      float mn = fmaxf(mr[r], rmax[r]);
      alpha[r] = EXP2F((mr[r] - mn) * LOG2E);
      p0[r]    = EXP2F((S0[r] - mn) * LOG2E);
      p1[r]    = EXP2F((S1[r] - mn) * LOG2E);
      mr[r]    = mn;
      rsum[r]  = p0[r] + p1[r];
    }
#pragma unroll
    for (int m = 1; m < 16; m <<= 1)
#pragma unroll
      for (int r = 0; r < 8; ++r) rsum[r] += __shfl_xor(rsum[r], m, 32);
#pragma unroll
    for (int r = 0; r < 8; ++r) lr[r] = lr[r] * alpha[r] + rsum[r];

    // ---- P: C-layout -> A-layout via per-wave LDS (wave-synchronous) ----
    unsigned short* pw = PL + wave * 512;
#pragma unroll
    for (int r = 0; r < 8; ++r) {
      const int row = r + 8 * lhi;
      pw[row * 32 + ln]      = f2bf(p0[r]);
      pw[row * 32 + 16 + ln] = f2bf(p1[r]);
    }
    Frag pa;
    pa.u8[0] = *(const ushort8*)(pw + ln * 32 + lhi * 16);
    pa.u8[1] = *(const ushort8*)(pw + ln * 32 + lhi * 16 + 8);

    // ---- O = O*alpha + P.V (8 accumulating WMMAs) ----
    const unsigned short* vt = VF + buf * 8192 + (wc * 8) * 512;
#pragma unroll
    for (int ct = 0; ct < 8; ++ct) {
#pragma unroll
      for (int r = 0; r < 8; ++r) O[ct][r] *= alpha[r];
      Frag vf;
      const unsigned short* vp = vt + ct * 512 + lane * 16;
      vf.u8[0] = *(const ushort8*)vp;
      vf.u8[1] = *(const ushort8*)(vp + 8);
      O[ct] = __builtin_amdgcn_wmma_f32_16x16x32_bf16(
          false, pa.v, false, vf.v, (short)0, O[ct], false, false);
    }

    asm volatile("s_wait_asynccnt 0" ::: "memory");
    __syncthreads();
  }

  // ---- epilogue: O/l -> LDS transpose (stride 133) -> coalesced out + x ----
  float invl[8];
#pragma unroll
  for (int r = 0; r < 8; ++r) invl[r] = 1.0f / lr[r];

  float* OL = (float*)smem;  // [64][133], reuses VF/PL space
  for (int h = 0; h < 2; ++h) {
    __syncthreads();
    if (wc == h) {
#pragma unroll
      for (int ct = 0; ct < 8; ++ct)
#pragma unroll
        for (int r = 0; r < 8; ++r)
          OL[(wi * 16 + r + 8 * lhi) * 133 + ct * 16 + ln] = O[ct][r] * invl[r];
    }
    __syncthreads();
    const int i  = tid & 63;
    const int cb = tid >> 6;
    for (int rep = 0; rep < 32; ++rep) {
      const int c = h * 128 + cb + rep * 4;
      const size_t gidx = (size_t)(b * 256 + c) * 4096 + i0 + i;
      out[gidx] = OL[i * 133 + (cb + rep * 4)] + x[gidx];
    }
  }
}

// ---------------------------------------------------------------------------
extern "C" void kernel_launch(void* const* d_in, const int* in_sizes, int n_in,
                              void* d_out, int out_size, void* d_ws,
                              size_t ws_size, hipStream_t stream) {
  (void)in_sizes; (void)n_in; (void)out_size; (void)ws_size;
  const float* x  = (const float*)d_in[0];
  const float* Wq = (const float*)d_in[1];
  const float* bq = (const float*)d_in[2];
  const float* Wk = (const float*)d_in[3];
  const float* bk = (const float*)d_in[4];
  const float* Wv = (const float*)d_in[5];
  const float* bv = (const float*)d_in[6];
  float* out = (float*)d_out;

  char* ws = (char*)d_ws;
  unsigned short* Qws  = (unsigned short*)(ws);                         // 1 MiB
  unsigned short* Kws  = (unsigned short*)(ws + (1u << 20));            // 1 MiB
  unsigned short* Vg   = (unsigned short*)(ws + (2u << 20));            // 8 MiB
  unsigned short* Wall = (unsigned short*)(ws + (10u << 20));           // 160 KiB
  float*          ball = (float*)(ws + (10u << 20) + 320 * 256 * 2);    // 1.25 KiB

  k_prep<<<320, 256, 0, stream>>>(Wq, Wk, Wv, bq, bk, bv, Wall, ball);
  k_proj<<<dim3(64, 4), 256, 0, stream>>>(x, Wall, ball, Qws, Kws, Vg);
  k_attn<<<dim3(64, 4), 256, 0, stream>>>(x, Qws, Kws, Vg, out);
}
